// MultiheadAttention_73023033966797
// MI455X (gfx1250) — compile-verified, matrix operand order checked by reading, not run on hardware
//
#include <hip/hip_runtime.h>

typedef __attribute__((ext_vector_type(16))) __bf16 v16bf;
typedef __attribute__((ext_vector_type(8)))  float  v8f;
typedef __attribute__((ext_vector_type(16))) unsigned short u16x16;
typedef __attribute__((ext_vector_type(8)))  unsigned short u16x8;

#define EMBED 768
#define HEADS 32
#define HDIM  24
#define HPAD  32
#define TSEQ  512
#define BATCH 16
#define ROWS  (TSEQ * BATCH)   // 8192

// ---------------------------------------------------------------- helpers

__device__ __forceinline__ unsigned short f32_bf16(float f) {
  unsigned int u = __builtin_bit_cast(unsigned int, f);
  unsigned int r = u + 0x7FFFu + ((u >> 16) & 1u);   // round-to-nearest-even
  return (unsigned short)(r >> 16);
}

// Load a 16x32 (outer x K) bf16 fragment in the CDNA5 16-bit A/B layout:
// lane l<16 : outer = row0+l, K = k0+0..7 and k0+16..23
// lane l>=16: outer = row0+(l-16), K = k0+8..15 and k0+24..31
__device__ __forceinline__ v16bf load_frag(const unsigned short* __restrict__ base,
                                           int ld, int row0, int k0, int lane) {
  const int l = lane & 15, hi = lane >> 4;
  const unsigned short* p = base + (size_t)(row0 + l) * ld + k0 + hi * 8;
  u16x8 a = *(const u16x8*)(p);
  u16x8 b = *(const u16x8*)(p + 16);
  union { u16x8 h[2]; u16x16 v; } u;
  u.h[0] = a; u.h[1] = b;
  return __builtin_bit_cast(v16bf, u.v);
}

__device__ __forceinline__ v8f wmma_bf16(v16bf a, v16bf b, v8f c) {
  return __builtin_amdgcn_wmma_f32_16x16x32_bf16(
      /*neg_a=*/false, a, /*neg_b=*/false, b,
      /*c_mod=*/(short)0, c, /*reuse_a=*/false, /*reuse_b=*/false);
}

// ---------------------------------------------------------------- utility kernels

__global__ void cvt_f32_bf16(const float* __restrict__ in,
                             unsigned short* __restrict__ out, int n) {
  int i = blockIdx.x * blockDim.x + threadIdx.x;
  if (i < n) out[i] = f32_bf16(in[i]);
}

__global__ void zero_u16(unsigned short* __restrict__ p, int n) {
  int i = blockIdx.x * blockDim.x + threadIdx.x;
  if (i < n) p[i] = 0;
}

// ---------------------------------------------------------------- projection GEMM
// out[m,n] = (sum_k X[m,k] * W[n,k] + bias[n]) * scale   (y = x @ W^T + b)
// 32x32 output tile per wave: 2x2 WMMA per k-step, each fragment reused twice.
// mode 0: -> Qpad bf16 [B*H][T][HPAD]   (scale = HDIM^-0.5)
// mode 1: -> Kpad bf16 [B*H][T][HPAD]
// mode 2: -> Vt   bf16 [B*H][HPAD][T]   (transposed)
// mode 3: -> fp32 [ROWS][EMBED]  (== [T,B,E] since m = t*B + b)

__global__ __launch_bounds__(256) void proj_gemm(
    const unsigned short* __restrict__ X, const unsigned short* __restrict__ W,
    const float* __restrict__ bias, unsigned short* __restrict__ outb,
    float* __restrict__ outf, int mode, float scale)
{
  const int lane = threadIdx.x & 31;
  const int wave = threadIdx.x >> 5;
  const int tile = blockIdx.x * 8 + wave;        // (8192/32) * (768/32) = 6144 tiles
  const int mt = tile / (EMBED / 32);
  const int nt = tile % (EMBED / 32);
  const int row0 = mt * 32, col0 = nt * 32;

  v8f acc00 = {}, acc01 = {}, acc10 = {}, acc11 = {};
  for (int k0 = 0; k0 < EMBED; k0 += 32) {
    v16bf a0 = load_frag(X, EMBED, row0,      k0, lane);
    v16bf a1 = load_frag(X, EMBED, row0 + 16, k0, lane);
    v16bf b0 = load_frag(W, EMBED, col0,      k0, lane);
    v16bf b1 = load_frag(W, EMBED, col0 + 16, k0, lane);
    acc00 = wmma_bf16(a0, b0, acc00);
    acc01 = wmma_bf16(a0, b1, acc01);
    acc10 = wmma_bf16(a1, b0, acc10);
    acc11 = wmma_bf16(a1, b1, acc11);
  }

  const int l = lane & 15, hi = lane >> 4;
  v8f accs[2][2] = { { acc00, acc01 }, { acc10, acc11 } };
#pragma unroll
  for (int ai = 0; ai < 2; ++ai) {
#pragma unroll
    for (int ni = 0; ni < 2; ++ni) {
      v8f acc = accs[ai][ni];
#pragma unroll
      for (int i = 0; i < 8; ++i) {
        const int m = row0 + ai * 16 + i + 8 * hi;   // C layout: vgpr i -> row i + 8*halfwave
        const int n = col0 + ni * 16 + l;            // col = lane % 16
        const float v = (acc[i] + bias[n]) * scale;
        if (mode == 3) {
          outf[(size_t)m * EMBED + n] = v;
        } else {
          const int t  = m >> 4;                     // m = t*BATCH + b  (BATCH == 16)
          const int bi = m & 15;
          const int h  = n / HDIM;
          const int d  = n % HDIM;
          const int bh = bi * HEADS + h;
          size_t idx;
          if (mode == 2) idx = ((size_t)bh * HPAD + d) * TSEQ + t;    // V transposed
          else           idx = ((size_t)bh * TSEQ + t) * HPAD + d;    // Q / K
          outb[idx] = f32_bf16(v);
        }
      }
    }
  }
}

// ---------------------------------------------------------------- fused flash attention
// One wave per (bh, 16-row tile). Streams keys in chunks of 32 with online softmax;
// reads the 536 MB attn_bias exactly once (the HBM roofline floor for this op).

__global__ __launch_bounds__(32) void attn_kernel(
    const unsigned short* __restrict__ Qp, const unsigned short* __restrict__ Kp,
    const unsigned short* __restrict__ Vt, const float* __restrict__ bias,
    unsigned short* __restrict__ ctx)
{
  __shared__ unsigned short ldsP[16 * HPAD];     // 16x32 bf16 P tile (1 KB)
  const int lane = threadIdx.x;
  const int bh = blockIdx.x >> 5;                // T/16 == 32 row tiles
  const int rt = blockIdx.x & 31;
  const int l = lane & 15, hi = lane >> 4;

  const unsigned short* Qb = Qp + (size_t)bh * TSEQ * HPAD;
  const unsigned short* Kb = Kp + (size_t)bh * TSEQ * HPAD;
  const unsigned short* Vb = Vt + (size_t)bh * HPAD * TSEQ;
  const float* Bb = bias + (size_t)bh * TSEQ * TSEQ;

  const v16bf a_q = load_frag(Qb, HPAD, rt * 16, 0, lane);   // Q rows, K-dim = padded 32

  v8f acc0 = {}, acc1 = {};                      // context dims 0..15 / 16..31
  float mrow[8], lrow[8];
#pragma unroll
  for (int i = 0; i < 8; ++i) { mrow[i] = -3.0e38f; lrow[i] = 0.f; }

  for (int s0 = 0; s0 < TSEQ; s0 += 32) {
    v16bf bk0 = load_frag(Kb, HPAD, s0,      0, lane);
    v16bf bk1 = load_frag(Kb, HPAD, s0 + 16, 0, lane);
    v8f zf = {};
    v8f sc0 = wmma_bf16(a_q, bk0, zf);           // S tile cols s0..s0+15
    v8f sc1 = wmma_bf16(a_q, bk1, zf);           // S tile cols s0+16..s0+31

#pragma unroll
    for (int i = 0; i < 8; ++i) {                // bias add in C-fragment layout
      const int trow = rt * 16 + i + 8 * hi;
      sc0[i] += Bb[(size_t)trow * TSEQ + s0 + l];
      sc1[i] += Bb[(size_t)trow * TSEQ + s0 + 16 + l];
    }

#pragma unroll
    for (int i = 0; i < 8; ++i) {                // online softmax (rows across 16 lanes)
      float mx = fmaxf(sc0[i], sc1[i]);
#pragma unroll
      for (int off = 1; off < 16; off <<= 1) mx = fmaxf(mx, __shfl_xor(mx, off, 32));
      const float mnew = fmaxf(mrow[i], mx);
      const float p0 = __expf(sc0[i] - mnew);
      const float p1 = __expf(sc1[i] - mnew);
      float rs = p0 + p1;
#pragma unroll
      for (int off = 1; off < 16; off <<= 1) rs += __shfl_xor(rs, off, 32);
      const float alpha = __expf(mrow[i] - mnew);
      lrow[i] = lrow[i] * alpha + rs;
      mrow[i] = mnew;
      sc0[i] = p0; sc1[i] = p1;
      acc0[i] *= alpha; acc1[i] *= alpha;
    }

    // stage P through LDS: C-fragment layout -> 16-bit A-fragment layout
#pragma unroll
    for (int i = 0; i < 8; ++i) {
      const int r = i + 8 * hi;
      ldsP[r * HPAD + l]      = f32_bf16(sc0[i]);
      ldsP[r * HPAD + 16 + l] = f32_bf16(sc1[i]);
    }
    __syncthreads();

    const v16bf a_p = load_frag(ldsP, HPAD, 0, 0, lane);       // P [16 x 32 keys]
    const v16bf bv0 = load_frag(Vb, TSEQ, 0,  s0, lane);       // V^T rows = dims 0..15
    const v16bf bv1 = load_frag(Vb, TSEQ, 16, s0, lane);       // dims 16..31 (24..31 zero)
    acc0 = wmma_bf16(a_p, bv0, acc0);
    acc1 = wmma_bf16(a_p, bv1, acc1);
    __syncthreads();
  }

  const int b = bh >> 5;                          // bh = b*HEADS + h, HEADS == 32
  const int h = bh & 31;
#pragma unroll
  for (int i = 0; i < 8; ++i) {
    const float inv = 1.0f / lrow[i];
    const int t = rt * 16 + i + 8 * hi;
    const size_t rowbase = ((size_t)t * BATCH + b) * EMBED + (size_t)h * HDIM;
    ctx[rowbase + l] = f32_bf16(acc0[i] * inv);
    if (16 + l < HDIM) ctx[rowbase + 16 + l] = f32_bf16(acc1[i] * inv);
  }
}

// ---------------------------------------------------------------- launcher

extern "C" void kernel_launch(void* const* d_in, const int* in_sizes, int n_in,
                              void* d_out, int out_size, void* d_ws, size_t ws_size,
                              hipStream_t stream) {
  const float* query = (const float*)d_in[0];
  const float* abias = (const float*)d_in[1];
  const float* Wq = (const float*)d_in[2];
  const float* bq = (const float*)d_in[3];
  const float* Wk = (const float*)d_in[4];
  const float* bk = (const float*)d_in[5];
  const float* Wv = (const float*)d_in[6];
  const float* bv = (const float*)d_in[7];
  const float* Wo = (const float*)d_in[8];
  const float* bo = (const float*)d_in[9];
  float* out = (float*)d_out;

  char* ws = (char*)d_ws;
  size_t off = 0;
  auto alloc = [&](size_t bytes) -> void* {
    void* p = ws + off;
    off += (bytes + 255) & ~(size_t)255;
    return p;
  };

  const size_t nX  = (size_t)ROWS * EMBED;                 // 6,291,456
  const size_t nW  = (size_t)EMBED * EMBED;                // 589,824
  const size_t nHP = (size_t)HEADS * BATCH * TSEQ * HPAD;  // 8,388,608 per tensor

  unsigned short* Xbf = (unsigned short*)alloc(nX * 2);
  unsigned short* Wqb = (unsigned short*)alloc(nW * 2);
  unsigned short* Wkb = (unsigned short*)alloc(nW * 2);
  unsigned short* Wvb = (unsigned short*)alloc(nW * 2);
  unsigned short* Wob = (unsigned short*)alloc(nW * 2);
  unsigned short* Qp  = (unsigned short*)alloc(nHP * 2);
  unsigned short* Kp  = (unsigned short*)alloc(nHP * 2);
  unsigned short* Vtp = (unsigned short*)alloc(nHP * 2);
  unsigned short* ctx = (unsigned short*)alloc(nX * 2);
  (void)ws_size; (void)in_sizes; (void)n_in; (void)out_size;

  // fp32 -> bf16 conversions
  cvt_f32_bf16<<<(int)((nX + 255) / 256), 256, 0, stream>>>(query, Xbf, (int)nX);
  cvt_f32_bf16<<<(int)((nW + 255) / 256), 256, 0, stream>>>(Wq, Wqb, (int)nW);
  cvt_f32_bf16<<<(int)((nW + 255) / 256), 256, 0, stream>>>(Wk, Wkb, (int)nW);
  cvt_f32_bf16<<<(int)((nW + 255) / 256), 256, 0, stream>>>(Wv, Wvb, (int)nW);
  cvt_f32_bf16<<<(int)((nW + 255) / 256), 256, 0, stream>>>(Wo, Wob, (int)nW);

  // zero padded Q/K/Vt (contiguous; pads d=24..31 so WMMA K=32 is exact)
  const size_t nPadAll = nHP * 3;
  zero_u16<<<(int)((nPadAll + 255) / 256), 256, 0, stream>>>(Qp, (int)nPadAll);

  // projections: (8192/32) x (768/32) = 6144 wave-tiles, 8 waves per 256-thread block
  const int gemmBlocks = (ROWS / 32) * (EMBED / 32) / 8;   // 768
  const float scaleQ = 0.20412414523193154f;               // HDIM^-0.5
  proj_gemm<<<gemmBlocks, 256, 0, stream>>>(Xbf, Wqb, bq, Qp,  nullptr, 0, scaleQ);
  proj_gemm<<<gemmBlocks, 256, 0, stream>>>(Xbf, Wkb, bk, Kp,  nullptr, 1, 1.0f);
  proj_gemm<<<gemmBlocks, 256, 0, stream>>>(Xbf, Wvb, bv, Vtp, nullptr, 2, 1.0f);

  // fused attention: one wave per (bh, 16-row tile): 512 * 32 = 16384 blocks
  attn_kernel<<<HEADS * BATCH * (TSEQ / 16), 32, 0, stream>>>(Qp, Kp, Vtp, abias, ctx);

  // output projection -> fp32 d_out
  proj_gemm<<<gemmBlocks, 256, 0, stream>>>(ctx, Wob, bo, nullptr, out, 3, 1.0f);
}
